// MultiHeadAttention_8375186227937
// MI455X (gfx1250) — matrix operand order repaired and checked by reading, not compiled, not run on hardware
//
#include <hip/hip_runtime.h>

#define EDIM 1024
#define HEADS 16
#define DHEAD 64
#define BATCH 8
#define SEQ 1024
#define NTOK (BATCH * SEQ)   // 8192

typedef __attribute__((ext_vector_type(16))) _Float16 v16h;
typedef __attribute__((ext_vector_type(8)))  _Float16 v8h;
typedef __attribute__((ext_vector_type(8)))  float    v8f;
typedef __attribute__((ext_vector_type(4)))  unsigned int u32x4;
typedef __attribute__((ext_vector_type(8)))  int          i32x8;
typedef __attribute__((ext_vector_type(4)))  int          i32x4;

#if __has_builtin(__builtin_amdgcn_tensor_load_to_lds) && __has_builtin(__builtin_amdgcn_s_wait_tensorcnt)
#define HAVE_TDM 1
#else
#define HAVE_TDM 0
#endif

// ---- WMMA fragment helpers (ISA 7.12.2 layouts, wave32) -------------------
// A (16x32 f16): lane m=lane&15 holds row m; lanes 0-15 take K {0..7,16..23},
// lanes 16-31 take K {8..15,24..31}.  hi8 = (lane>=16) ? 8 : 0.
static __device__ __forceinline__ v16h load_a_frag(const _Float16* __restrict__ p, int hi8) {
  const _Float16* q = p + hi8;
  v8h lo = *(const v8h*)(q);
  v8h hi = *(const v8h*)(q + 16);
  v16h r;
#pragma unroll
  for (int i = 0; i < 8; ++i) { r[i] = lo[i]; r[i + 8] = hi[i]; }
  return r;
}

// B (32x16 f16): lane n=lane&15 holds column n; lanes 0-15 take K {0..7,16..23},
// lanes 16-31 take K {8..15,24..31} (same as A).  hi16 = (lane>=16) ? 16 : 0.
static __device__ __forceinline__ v16h load_b_frag(const _Float16* __restrict__ p, int hi16) {
  const _Float16* q = p + (hi16 >> 1);
  v8h lo = *(const v8h*)(q);
  v8h hi = *(const v8h*)(q + 16);
  v16h r;
#pragma unroll
  for (int i = 0; i < 8; ++i) { r[i] = lo[i]; r[i + 8] = hi[i]; }
  return r;
}

static __device__ __forceinline__ v8f wmma_f16(v16h a, v16h b, v8f c) {
  return __builtin_amdgcn_wmma_f32_16x16x32_f16(false, a, false, b, (short)0, c, false, false);
}

// ---- TDM: 2D f16 tile (rows x 64 elems, row stride in elems) -> LDS -------
// D# layout per CDNA5 ISA (group0 128b, group1 256b).
#if HAVE_TDM
static __device__ __forceinline__ void tdm_load_k_tile(const _Float16* g, unsigned lds_off,
                                                       int rows, int stride_elts) {
  unsigned long long ga = (unsigned long long)(uintptr_t)g;
  u32x4 g0;
  g0[0] = 1u;                                            // count=1, no gather
  g0[1] = lds_off;                                       // lds_addr (bytes)
  g0[2] = (unsigned)(ga & 0xffffffffu);                  // global_addr[31:0]
  g0[3] = (unsigned)((ga >> 32) & 0x01ffffffu) | (2u << 30); // addr[56:32] | type=2
  i32x8 g1;
  g1[0] = 1 << 16;                                       // data_size=1 (2 bytes)
  g1[1] = (64 & 0xffff) << 16;                           // tensor_dim0 = 64 (lo16)
  g1[2] = ((rows & 0xffff) << 16);                       // dim0 hi=0 | tensor_dim1 lo
  g1[3] = (64 & 0xffff) << 16;                           // dim1 hi=0 | tile_dim0 = 64
  g1[4] = rows & 0xffff;                                 // tile_dim1 (tile_dim2=0)
  g1[5] = stride_elts;                                   // tensor_dim0_stride lo32
  g1[6] = 0;
  g1[7] = 0;
  i32x4 zz = {0, 0, 0, 0};                               // groups 2/3 unused (2D)
  i32x8 z8 = {0, 0, 0, 0, 0, 0, 0, 0};                   // extra group (6-arg form)
  __builtin_amdgcn_tensor_load_to_lds(g0, g1, zz, zz, z8, 0);
}
#endif

// stage 32x64 f16 tile into LDS (TDM if available, else cooperative copy)
static __device__ __forceinline__ void k_tile_issue(const _Float16* g, _Float16* l,
                                                    unsigned lds_off, int lane) {
#if HAVE_TDM
  (void)l; (void)lane;
  tdm_load_k_tile(g, lds_off, 32, EDIM);
#else
  (void)lds_off;
#pragma unroll 4
  for (int r = lane; r < 32 * 8; r += 32) {   // 8 x v8h per 64-elem row
    int row = r >> 3, c = (r & 7) * 8;
    *(v8h*)(l + row * 64 + c) = *(const v8h*)(g + (size_t)row * EDIM + c);
  }
#endif
}

// ---- utility kernels ------------------------------------------------------
__global__ void cast_f32_to_f16(const float* __restrict__ src, _Float16* __restrict__ dst, int n) {
  int i = blockIdx.x * blockDim.x + threadIdx.x;
  if (i < n) dst[i] = (_Float16)src[i];
}

__global__ void zero_f32(float* __restrict__ p, int n) {
  int i = blockIdx.x * blockDim.x + threadIdx.x;
  if (i < n) p[i] = 0.0f;
}

// ---- NT GEMM: out = (X @ W^T + bias) * scale, 32x64 tile per wave ---------
// mode 0: f16 row-major out; mode 1: f16 Vt[b,h,d,s]; mode 2: f32 row-major out.
__global__ __launch_bounds__(32) void gemm_nt(
    const _Float16* __restrict__ X, const _Float16* __restrict__ W,
    const float* __restrict__ bias, float scale,
    _Float16* __restrict__ outh, float* __restrict__ outf, int mode)
{
  const int lane = threadIdx.x & 31;
  const int n16  = lane & 15;
  const int hi8  = (lane >> 4) * 8;
  const int hi16 = (lane >> 4) * 16;
  const int mtile = blockIdx.x;   // M/32 tiles
  const int nt4   = blockIdx.y;   // EDIM/64 tiles

  const _Float16* xrow0 = X + (size_t)(mtile * 32 + n16) * EDIM;
  const _Float16* xrow1 = xrow0 + (size_t)16 * EDIM;
  v8f acc[2][4] = {};
  for (int k0 = 0; k0 < EDIM; k0 += 32) {
    v16h a0 = load_a_frag(xrow0 + k0, hi8);
    v16h a1 = load_a_frag(xrow1 + k0, hi8);
#pragma unroll
    for (int j = 0; j < 4; ++j) {
      const int ncol = nt4 * 64 + j * 16 + n16;
      v16h bfr = load_b_frag(W + (size_t)ncol * EDIM + k0, hi16);
      acc[0][j] = wmma_f16(a0, bfr, acc[0][j]);
      acc[1][j] = wmma_f16(a1, bfr, acc[1][j]);
    }
  }
#pragma unroll
  for (int i = 0; i < 2; ++i) {
    const int rbase = mtile * 32 + i * 16 + hi8;
#pragma unroll
    for (int j = 0; j < 4; ++j) {
      const int ncol = nt4 * 64 + j * 16 + n16;
      const float bb = bias[ncol];
#pragma unroll
      for (int r = 0; r < 8; ++r) {
        float v = (acc[i][j][r] + bb) * scale;
        const int row = rbase + r;
        if (mode == 0) {
          outh[(size_t)row * EDIM + ncol] = (_Float16)v;
        } else if (mode == 1) {           // transposed V: Vt[((b*H+h)*D+d)*S + s]
          const int bb_ = row >> 10, s = row & 1023;
          const int h = ncol >> 6, d = ncol & 63;
          outh[(((size_t)(bb_ * HEADS + h)) * DHEAD + d) * SEQ + s] = (_Float16)v;
        } else {
          outf[(size_t)row * EDIM + ncol] = v;
        }
      }
    }
  }
}

// ---- fused attention: stats pass + (P, attn_max, O = P@V) pass ------------
__global__ __launch_bounds__(32) void attn_fused(
    const _Float16* __restrict__ Qh, const _Float16* __restrict__ Kh,
    const _Float16* __restrict__ Vt,
    _Float16* __restrict__ rep, float* __restrict__ amax)
{
  __shared__ _Float16 ldsK[2][32 * 64];   // double-buffered K tiles (TDM target)
  __shared__ _Float16 ldsP[16 * 32];      // P tile staging: C-layout -> A-layout

  const int lane = threadIdx.x & 31;
  const int n16  = lane & 15;
  const int hi8  = (lane >> 4) * 8;
  const int hi16 = (lane >> 4) * 16;
  const int wid  = blockIdx.x;            // (b*H + h)*64 + ttile
  const int ttile = wid & 63;
  const int bh = wid >> 6;
  const int h = bh & (HEADS - 1);
  const int b = bh >> 4;

  const _Float16* qbase = Qh + ((size_t)(b * SEQ) + ttile * 16 + n16) * EDIM + h * DHEAD;
  v16h a0 = load_a_frag(qbase, hi8);        // d = 0..31
  v16h a1 = load_a_frag(qbase + 32, hi8);   // d = 32..63

  const _Float16* kpanel = Kh + (size_t)(b * SEQ) * EDIM + h * DHEAD;
  const unsigned lOff0 = (unsigned)(uintptr_t)&ldsK[0][0];
  const unsigned lOff1 = (unsigned)(uintptr_t)&ldsK[1][0];

  float m[8], z[8];
#pragma unroll
  for (int r = 0; r < 8; ++r) { m[r] = -1e30f; z[r] = 0.0f; }

  // ---------- pass 1: per-row (m, Z), rows stay in registers ----------
  k_tile_issue(kpanel, &ldsK[0][0], lOff0, lane);
  for (int sp = 0; sp < SEQ / 32; ++sp) {
#if HAVE_TDM
    if (sp + 1 < SEQ / 32) {
      k_tile_issue(kpanel + (size_t)(sp + 1) * 32 * EDIM, &ldsK[(sp + 1) & 1][0],
                   ((sp + 1) & 1) ? lOff1 : lOff0, lane);
      __builtin_amdgcn_s_wait_tensorcnt(1);   // oldest (current tile) done
    } else {
      __builtin_amdgcn_s_wait_tensorcnt(0);
    }
    const _Float16* kt = &ldsK[sp & 1][0];
#else
    if (sp > 0) k_tile_issue(kpanel + (size_t)sp * 32 * EDIM, &ldsK[0][0], lOff0, lane);
    const _Float16* kt = &ldsK[0][0];
#endif
#pragma unroll
    for (int t2 = 0; t2 < 2; ++t2) {
      const _Float16* kb = kt + (t2 * 16 + n16) * 64;
      v16h b0 = load_b_frag(kb, hi16);
      v16h b1 = load_b_frag(kb + 32, hi16);
      v8f c = {};
      c = wmma_f16(a0, b0, c);
      c = wmma_f16(a1, b1, c);
#pragma unroll
      for (int r = 0; r < 8; ++r) {
        float v = c[r];
        float tmax = v;
        tmax = fmaxf(tmax, __shfl_xor(tmax, 1, 32));
        tmax = fmaxf(tmax, __shfl_xor(tmax, 2, 32));
        tmax = fmaxf(tmax, __shfl_xor(tmax, 4, 32));
        tmax = fmaxf(tmax, __shfl_xor(tmax, 8, 32));
        float e = __expf(v - tmax);
        float tsum = e;
        tsum += __shfl_xor(tsum, 1, 32);
        tsum += __shfl_xor(tsum, 2, 32);
        tsum += __shfl_xor(tsum, 4, 32);
        tsum += __shfl_xor(tsum, 8, 32);
        float mn = fmaxf(m[r], tmax);
        z[r] = z[r] * __expf(m[r] - mn) + tsum * __expf(tmax - mn);
        m[r] = mn;
      }
    }
  }
  float zinv[8];
#pragma unroll
  for (int r = 0; r < 8; ++r) zinv[r] = 1.0f / z[r];

  // ---------- pass 2: p = exp(s-m)/Z, head-max reduction, O += P@V ----------
  v8f o[4] = {};
  k_tile_issue(kpanel, &ldsK[0][0], lOff0, lane);
  for (int sp = 0; sp < SEQ / 32; ++sp) {
#if HAVE_TDM
    if (sp + 1 < SEQ / 32) {
      k_tile_issue(kpanel + (size_t)(sp + 1) * 32 * EDIM, &ldsK[(sp + 1) & 1][0],
                   ((sp + 1) & 1) ? lOff1 : lOff0, lane);
      __builtin_amdgcn_s_wait_tensorcnt(1);
    } else {
      __builtin_amdgcn_s_wait_tensorcnt(0);
    }
    const _Float16* kt = &ldsK[sp & 1][0];
#else
    if (sp > 0) k_tile_issue(kpanel + (size_t)sp * 32 * EDIM, &ldsK[0][0], lOff0, lane);
    const _Float16* kt = &ldsK[0][0];
#endif
#pragma unroll
    for (int t2 = 0; t2 < 2; ++t2) {
      const int scol = sp * 32 + t2 * 16 + n16;
      const _Float16* kb = kt + (t2 * 16 + n16) * 64;
      v16h b0 = load_b_frag(kb, hi16);
      v16h b1 = load_b_frag(kb + 32, hi16);
      v8f c = {};
      c = wmma_f16(a0, b0, c);
      c = wmma_f16(a1, b1, c);
#pragma unroll
      for (int r = 0; r < 8; ++r) {
        float pv = __expf(c[r] - m[r]) * zinv[r];
        // max over heads: p > 0 so integer max on float bits is order-preserving
        atomicMax((int*)(amax + ((size_t)(b * SEQ) + ttile * 16 + hi8 + r) * SEQ + scol),
                  __float_as_int(pv));
        ldsP[(hi8 + r) * 32 + t2 * 16 + n16] = (_Float16)pv;
      }
    }
    // reload P in A-fragment layout (in-wave DS ordering guarantees visibility)
    v16h ap = load_a_frag(&ldsP[n16 * 32], hi8);
#pragma unroll
    for (int j = 0; j < 4; ++j) {
      const _Float16* vb = Vt + ((size_t)(bh * DHEAD) + j * 16 + n16) * SEQ + sp * 32;
      v16h bv = load_b_frag(vb, hi16);
      o[j] = wmma_f16(ap, bv, o[j]);
    }
  }
#pragma unroll
  for (int j = 0; j < 4; ++j)
#pragma unroll
    for (int r = 0; r < 8; ++r)
      rep[((size_t)(b * SEQ) + ttile * 16 + hi8 + r) * EDIM + h * DHEAD + j * 16 + n16] =
          (_Float16)o[j][r];
}

// ---- host-side orchestration ----------------------------------------------
extern "C" void kernel_launch(void* const* d_in, const int* in_sizes, int n_in,
                              void* d_out, int out_size, void* d_ws, size_t ws_size,
                              hipStream_t stream) {
  const float* query = (const float*)d_in[0];
  const float* key   = (const float*)d_in[1];
  const float* value = (const float*)d_in[2];
  const float* Wq = (const float*)d_in[3];  const float* bq = (const float*)d_in[4];
  const float* Wk = (const float*)d_in[5];  const float* bk = (const float*)d_in[6];
  const float* Wv = (const float*)d_in[7];  const float* bv = (const float*)d_in[8];
  const float* Wo = (const float*)d_in[9];  const float* bo = (const float*)d_in[10];

  constexpr size_t SZ_TOK16 = (size_t)NTOK * EDIM * 2;   // 16 MiB
  constexpr size_t SZ_W16   = (size_t)EDIM * EDIM * 2;   //  2 MiB

  char* ws = (char*)d_ws;
  size_t off = 0;
  auto take = [&](size_t bytes) { char* p = ws + off; off += (bytes + 255) & ~(size_t)255; return p; };
  _Float16* Xq   = (_Float16*)take(SZ_TOK16);
  _Float16* Xk   = (_Float16*)take(SZ_TOK16);
  _Float16* Xv   = (_Float16*)take(SZ_TOK16);
  _Float16* Qh   = (_Float16*)take(SZ_TOK16);
  _Float16* Kh   = (_Float16*)take(SZ_TOK16);
  _Float16* Vt   = (_Float16*)take(SZ_TOK16);
  _Float16* repb = (_Float16*)take(SZ_TOK16);
  _Float16* Wq16 = (_Float16*)take(SZ_W16);
  _Float16* Wk16 = (_Float16*)take(SZ_W16);
  _Float16* Wv16 = (_Float16*)take(SZ_W16);
  _Float16* Wo16 = (_Float16*)take(SZ_W16);

  float* out_f  = (float*)d_out;                       // [B,T,E]
  float* amax   = (float*)d_out + (size_t)NTOK * EDIM; // [B,T,S]

  const int nTok = NTOK * EDIM;
  const int nW   = EDIM * EDIM;
  cast_f32_to_f16<<<(nTok + 255) / 256, 256, 0, stream>>>(query, Xq, nTok);
  cast_f32_to_f16<<<(nTok + 255) / 256, 256, 0, stream>>>(key,   Xk, nTok);
  cast_f32_to_f16<<<(nTok + 255) / 256, 256, 0, stream>>>(value, Xv, nTok);
  cast_f32_to_f16<<<(nW + 255) / 256, 256, 0, stream>>>(Wq, Wq16, nW);
  cast_f32_to_f16<<<(nW + 255) / 256, 256, 0, stream>>>(Wk, Wk16, nW);
  cast_f32_to_f16<<<(nW + 255) / 256, 256, 0, stream>>>(Wv, Wv16, nW);
  cast_f32_to_f16<<<(nW + 255) / 256, 256, 0, stream>>>(Wo, Wo16, nW);

  const int nAmax = NTOK * SEQ;
  zero_f32<<<(nAmax + 255) / 256, 256, 0, stream>>>(amax, nAmax);

  dim3 ggrid(NTOK / 32, EDIM / 64);
  const float scaling = 0.125f;   // D^-0.5, D = 64
  gemm_nt<<<ggrid, 32, 0, stream>>>(Xq, Wq16, bq, scaling, Qh, nullptr, 0);
  gemm_nt<<<ggrid, 32, 0, stream>>>(Xk, Wk16, bk, 1.0f,    Kh, nullptr, 0);
  gemm_nt<<<ggrid, 32, 0, stream>>>(Xv, Wv16, bv, 1.0f,    Vt, nullptr, 1);

  const int nAttnBlocks = BATCH * HEADS * (SEQ / 16);   // 8192 blocks, 1 wave each
  attn_fused<<<nAttnBlocks, 32, 0, stream>>>(Qh, Kh, Vt, repb, amax);

  gemm_nt<<<ggrid, 32, 0, stream>>>(repb, Wo16, bo, 1.0f, nullptr, out_f, 2);
}
